// TransE_Gate_att_49203145342989
// MI455X (gfx1250) — matrix operand order repaired and checked by reading, not compiled, not run on hardware
//
#include <hip/hip_runtime.h>
#include <stdint.h>

typedef __attribute__((ext_vector_type(16))) _Float16 v16h;
typedef __attribute__((ext_vector_type(8)))  float    v8f;
typedef __attribute__((ext_vector_type(4)))  float    f32x4;

union HFrag { v16h v; f32x4 q[2]; _Float16 h[16]; };
union CFrag { v8f  v; float f[8]; };

#define NE     10000
#define NATT   64
#define EMB    128
#define NB     8
#define NH     8
#define NTILES 625   // 10000 / 16

// ---------------- deterministic counter-based RNG (threefry-style mix) ------
__device__ __forceinline__ unsigned mix32(unsigned x) {
  x ^= x >> 16; x *= 0x7FEB352Du; x ^= x >> 15; x *= 0x846CA68Bu; x ^= x >> 16;
  return x;
}
__device__ __forceinline__ float u01(unsigned a, unsigned b, unsigned salt) {
  unsigned h = mix32(a * 0x9E3779B9u + b * 0x85EBCA6Bu + salt);
  return (float)(h >> 8) * (1.0f / 16777216.0f);
}
__device__ __forceinline__ float bern05(unsigned a, unsigned b, unsigned salt) {
  unsigned h = mix32(a * 0x27D4EB2Fu + b * 0x165667B1u + salt);
  return (float)(h & 1u);
}

// ---------------- K0: all small precomputes + the e1 branch -----------------
__global__ void __launch_bounds__(128) k0_precompute(
    const float* __restrict__ emb_e, const float* __restrict__ emb_rel,
    const float* __restrict__ emb_att, const float* __restrict__ lits,
    const float* __restrict__ lin_w, const float* __restrict__ lin_b,
    const float* __restrict__ in_w,  const float* __restrict__ in_b,
    const float* __restrict__ out_w, const float* __restrict__ out_b,
    const float* __restrict__ g1_gw, const float* __restrict__ g1_gb,
    const float* __restrict__ g1_g1w, const float* __restrict__ g1_g2w,
    const float* __restrict__ g1_bias,
    const float* __restrict__ g2_gw, const float* __restrict__ g2_g1w,
    const float* __restrict__ g2_g2w,
    const int* __restrict__ e1, const int* __restrict__ rel,
    float* __restrict__ wsA, float* __restrict__ wsc, float* __restrict__ wsbias2,
    float* __restrict__ wstvec, float* __restrict__ wse1g,
    _Float16* __restrict__ U16, _Float16* __restrict__ WgT, _Float16* __restrict__ WcT)
{
  const int tid = threadIdx.x;
  __shared__ float relat[NB][EMB];
  __shared__ float qh[NB][EMB];
  __shared__ float buf[64][EMB];     // Khat, then Vhat
  __shared__ float Al[NB][NH][64];
  __shared__ float cl[NB][NH];
  __shared__ float litb[64];
  __shared__ float w1[512];
  __shared__ float xl[256];

  // rel_att = tanh(rel_emb @ lin_w.T + lin_b)
  for (int b = 0; b < NB; ++b) {
    const int a = tid;
    const float* rrow = emb_rel + (size_t)rel[b] * EMB;
    float s = lin_b[a];
    for (int e = 0; e < EMB; ++e) s += rrow[e] * lin_w[a * EMB + e];
    relat[b][a] = tanhf(s);
  }
  __syncthreads();
  // qh = (rel_att @ wq.T + bq) * Dh^-0.5
  for (int b = 0; b < NB; ++b) {
    const int j = tid;
    float s = in_b[j];
    for (int a = 0; a < EMB; ++a) s += relat[b][a] * in_w[j * EMB + a];
    qh[b][j] = s * 0.25f;
  }
  // Khat = emb_att @ wk.T
  for (int s = 0; s < 64; ++s) {
    const int j = tid;
    const float* arow = emb_att + (size_t)s * EMB;
    const float* wrow = in_w + (size_t)(EMB + j) * EMB;
    float v = 0.f;
    for (int a = 0; a < EMB; ++a) v += arow[a] * wrow[a];
    buf[s][j] = v;
  }
  __syncthreads();
  // A[l,h,s] = qh_h . Khat_h ; c[l,h] = qh_h . bk_h
  for (int idx = tid; idx < NB * NH * 64; idx += 128) {
    const int b = idx >> 9, h = (idx >> 6) & 7, s = idx & 63;
    float v = 0.f;
    for (int d = 0; d < 16; ++d) v += qh[b][h * 16 + d] * buf[s][h * 16 + d];
    Al[b][h][s] = v; wsA[idx] = v;
  }
  if (tid < 64) {
    const int b = tid >> 3, h = tid & 7;
    float v = 0.f;
    for (int d = 0; d < 16; ++d) v += qh[b][h * 16 + d] * in_b[EMB + h * 16 + d];
    cl[b][h] = v; wsc[tid] = v;
  }
  __syncthreads();
  // Vhat = emb_att @ wv.T (overwrite buf)
  for (int s = 0; s < 64; ++s) {
    const int j = tid;
    const float* arow = emb_att + (size_t)s * EMB;
    const float* wrow = in_w + (size_t)(2 * EMB + j) * EMB;
    float v = 0.f;
    for (int a = 0; a < EMB; ++a) v += arow[a] * wrow[a];
    buf[s][j] = v;
  }
  __syncthreads();
  // U[(h*64+s), f] = sum_d out_w[f, 16h+d] * Vhat[s, 16h+d]  (f16)
  for (int idx = tid; idx < 512 * EMB; idx += 128) {
    const int k = idx >> 7, f = idx & 127;
    const int h = k >> 6, s = k & 63;
    float v = 0.f;
    for (int d = 0; d < 16; ++d) v += out_w[f * EMB + h * 16 + d] * buf[s][h * 16 + d];
    U16[idx] = (_Float16)v;
  }
  { const int f = tid;                          // bias2 = out_w@bv + out_b
    float v = out_b[f];
    for (int e = 0; e < EMB; ++e) v += out_w[f * EMB + e] * in_b[2 * EMB + e];
    wsbias2[f] = v; }
  // transposed f16 gate weights: WgT[k][f]=g2_gw[f][k]; WcT=[g2_g1w|g2_g2w]^T
  for (int idx = tid; idx < 256 * EMB; idx += 128) {
    const int k = idx >> 7, f = idx & 127;
    WgT[idx] = (_Float16)g2_gw[f * 256 + k];
    WcT[idx] = (_Float16)((k < 128) ? g2_g1w[f * EMB + k] : g2_g2w[f * EMB + (k - 128)]);
  }
  __threadfence();
  __syncthreads();
  // ---- e1 branch (tiny: B=8) ----
  for (int b = 0; b < NB; ++b) {
    if (tid < 64) litb[tid] = lits[(size_t)e1[b] * NATT + tid];
    __syncthreads();
    if (tid < NH) {
      const int h = tid;
      const float c = cl[b][h];
      float mx = -1e30f;
      for (int s = 0; s < 64; ++s) mx = fmaxf(mx, litb[s] * Al[b][h][s] + c);
      float sum = 0.f;
      for (int s = 0; s < 64; ++s) {
        float e = __expf(litb[s] * Al[b][h][s] + c - mx);
        w1[h * 64 + s] = e; sum += e;
      }
      const float inv = 1.0f / sum;
      for (int s = 0; s < 64; ++s) w1[h * 64 + s] = w1[h * 64 + s] * litb[s] * inv;
    }
    __syncthreads();
    { const int f = tid;                         // e1_num_lit, build x = [ent|lit]
      float acc = wsbias2[f];
      for (int k = 0; k < 512; ++k) acc += w1[k] * (float)U16[k * EMB + f];
      xl[128 + f] = acc;
      xl[f] = emb_e[(size_t)e1[b] * EMB + f]; }
    __syncthreads();
    { const int f = tid;                         // gate (g1 weights)
      float ge = g1_gb[f];
      for (int k = 0; k < 256; ++k) ge += g1_gw[f * 256 + k] * xl[k];
      ge = tanhf(ge);
      float gt = g1_bias[f];
      for (int e = 0; e < 128; ++e) gt += g1_g1w[f * EMB + e] * xl[e];
      for (int a = 0; a < 128; ++a) gt += g1_g2w[f * EMB + a] * xl[128 + a];
      gt = 1.0f / (1.0f + __expf(-gt));
      const float v = (1.0f - gt) * xl[f] + gt * ge;
      wse1g[b * EMB + f] = v;
      wstvec[b * EMB + f] = v + emb_rel[(size_t)rel[b] * EMB + f]; }
    __syncthreads();
  }
}

// ---------------- K1: per-tile softmax + (16x512)@(512x128) WMMA ------------
__global__ void __launch_bounds__(32) k1_attention(
    const float* __restrict__ lits, const float* __restrict__ wsA,
    const float* __restrict__ wsc, const float* __restrict__ wsbias2,
    const _Float16* __restrict__ U16, float* __restrict__ e2nl)
{
  const int tile = blockIdx.x, l = blockIdx.y, lane = threadIdx.x;
  const int n0 = tile * 16;
  __shared__ float lit[16][64];
  __shared__ float Al[NH][64];
  __shared__ float cl[NH];
  __shared__ alignas(16) _Float16 wl[16][512];

  for (int i = lane; i < NH * 64; i += 32) Al[i >> 6][i & 63] = wsA[l * 512 + i];
  if (lane < NH) cl[lane] = wsc[l * NH + lane];
  for (int i = lane; i < 16 * 64; i += 32) {
    const int m = i >> 6, s = i & 63;
    lit[m][s] = lits[(size_t)(n0 + m) * NATT + s];
  }
  __syncthreads();
  // w[m][h*64+s] = softmax_s(lit*A + c) * lit   (f16 in LDS)
  for (int p = lane; p < 128; p += 32) {
    const int m = p >> 3, h = p & 7;
    const float c = cl[h];
    float mx = -1e30f;
    for (int s = 0; s < 64; ++s) mx = fmaxf(mx, lit[m][s] * Al[h][s] + c);
    float sum = 0.f;
    for (int s = 0; s < 64; ++s) {
      const float e = __expf(lit[m][s] * Al[h][s] + c - mx);
      sum += e; wl[m][h * 64 + s] = (_Float16)e;
    }
    const float inv = 1.0f / sum;
    for (int s = 0; s < 64; ++s)
      wl[m][h * 64 + s] = (_Float16)((float)wl[m][h * 64 + s] * lit[m][s] * inv);
  }
  __syncthreads();
  const int hi = lane >> 4, lo = lane & 15;
  for (int nt = 0; nt < 8; ++nt) {
    CFrag acc;
    const float bv = wsbias2[nt * 16 + lo];
    #pragma unroll
    for (int r = 0; r < 8; ++r) acc.f[r] = bv;
    #pragma unroll
    for (int kc = 0; kc < 16; ++kc) {
      HFrag a, bm;
      const _Float16* ar = &wl[lo][kc * 32 + hi * 8];       // A frag: lane->M, j->K
      a.q[0] = *(const f32x4*)ar;
      a.q[1] = *(const f32x4*)(ar + 16);
      const _Float16* br = U16 + (size_t)(kc * 32 + hi * 8) * EMB + nt * 16 + lo; // B: lane->N, j->K
      #pragma unroll
      for (int j = 0; j < 8; ++j) { bm.h[j] = br[j * EMB]; bm.h[8 + j] = br[(16 + j) * EMB]; }
      if (kc + 1 < 16)
        __builtin_prefetch(U16 + (size_t)((kc + 1) * 32 + lane) * EMB + nt * 16, 0, 0);
      acc.v = __builtin_amdgcn_wmma_f32_16x16x32_f16(false, a.v, false, bm.v,
                                                     (short)0, acc.v, false, false);
    }
    #pragma unroll
    for (int r = 0; r < 8; ++r) {                           // C: M=r+8*hi, N=lo
      const int n = n0 + r + 8 * hi;
      e2nl[((size_t)l * NE + n) * EMB + nt * 16 + lo] = acc.f[r];
    }
  }
}

// ---------------- K2: gate GEMMs + L1 score + pred ---------------------------
__global__ void __launch_bounds__(32) k2_gate_score(
    const float* __restrict__ emb_e, const float* __restrict__ e2nl,
    const _Float16* __restrict__ WgT, const _Float16* __restrict__ WcT,
    const float* __restrict__ g2_gb, const float* __restrict__ g2_bias,
    const float* __restrict__ wstvec, const float* __restrict__ e2_multi,
    float* __restrict__ pred_out, float* __restrict__ Spart)
{
  const int tile = blockIdx.x, l = blockIdx.y, lane = threadIdx.x;
  const int n0 = tile * 16;
  __shared__ alignas(16) _Float16 X[16][256];
  __shared__ float diffs[16][128];
  __shared__ float contrib[16];
  for (int i = lane; i < 16 * 256; i += 32) {
    const int m = i >> 8, k = i & 255;
    const float v = (k < 128) ? emb_e[(size_t)(n0 + m) * EMB + k]
                              : e2nl[((size_t)l * NE + n0 + m) * EMB + (k - 128)];
    X[m][k] = (_Float16)v;
  }
  __syncthreads();
  const int hi = lane >> 4, lo = lane & 15;
  for (int nt = 0; nt < 8; ++nt) {
    const int f = nt * 16 + lo;
    CFrag g, cc;
    const float gb = g2_gb[f], cb = g2_bias[f];
    #pragma unroll
    for (int r = 0; r < 8; ++r) { g.f[r] = gb; cc.f[r] = cb; }
    #pragma unroll
    for (int kc = 0; kc < 8; ++kc) {
      HFrag a, bg, bc;
      const _Float16* ar = &X[lo][kc * 32 + hi * 8];
      a.q[0] = *(const f32x4*)ar;
      a.q[1] = *(const f32x4*)(ar + 16);
      const _Float16* bgp = WgT + (size_t)(kc * 32 + hi * 8) * EMB + nt * 16 + lo;
      const _Float16* bcp = WcT + (size_t)(kc * 32 + hi * 8) * EMB + nt * 16 + lo;
      #pragma unroll
      for (int j = 0; j < 8; ++j) {
        bg.h[j] = bgp[j * EMB]; bg.h[8 + j] = bgp[(16 + j) * EMB];
        bc.h[j] = bcp[j * EMB]; bc.h[8 + j] = bcp[(16 + j) * EMB];
      }
      g.v  = __builtin_amdgcn_wmma_f32_16x16x32_f16(false, a.v, false, bg.v,
                                                    (short)0, g.v, false, false);
      cc.v = __builtin_amdgcn_wmma_f32_16x16x32_f16(false, a.v, false, bc.v,
                                                    (short)0, cc.v, false, false);
    }
    const float tvf = wstvec[l * EMB + f];
    #pragma unroll
    for (int r = 0; r < 8; ++r) {
      const int m = r + 8 * hi;
      const float ge = tanhf(g.f[r]);
      const float gt = 1.0f / (1.0f + __expf(-cc.f[r]));
      const float xe = (float)X[m][f];
      const float emb2 = (1.0f - gt) * xe + gt * ge;
      diffs[m][f] = fabsf(tvf - emb2);
    }
  }
  __syncthreads();
  if (lane < 16) {
    float s = 0.f;
    for (int f = 0; f < 128; ++f) s += diffs[lane][f];
    const float score = 9.0f - s;                    // GAMMA
    const float p = 1.0f / (1.0f + __expf(-score));
    const int gn = n0 + lane;
    pred_out[(size_t)l * NE + gn] = p;
    contrib[lane] = p * e2_multi[(size_t)l * NE + gn];
  }
  __syncthreads();
  if (lane == 0) {
    float t = 0.f;
    for (int i = 0; i < 16; ++i) t += contrib[i];
    Spart[l * NTILES + tile] = t;                    // deterministic partial
  }
}

__global__ void __launch_bounds__(32) k2b_reduceS(const float* __restrict__ Spart,
                                                  float* __restrict__ S)
{
  const int lane = threadIdx.x;
  if (lane < NB) {
    float s = 0.f;
    for (int t = 0; t < NTILES; ++t) s += Spart[lane * NTILES + t];
    S[lane] = s;
  }
}

// ---------------- K3: recompute gate, accumulate pos/neg partials ------------
__global__ void __launch_bounds__(32) k3_loss_accum(
    const float* __restrict__ emb_e, const float* __restrict__ e2nl,
    const _Float16* __restrict__ WgT, const _Float16* __restrict__ WcT,
    const float* __restrict__ g2_gb, const float* __restrict__ g2_bias,
    const float* __restrict__ e2_multi, const float* __restrict__ pred,
    const float* __restrict__ S,
    float* __restrict__ posP, float* __restrict__ negP,
    float* __restrict__ posWP, float* __restrict__ negWP)
{
  const int tile = blockIdx.x, l = blockIdx.y, lane = threadIdx.x;
  const int n0 = tile * 16;
  __shared__ alignas(16) _Float16 X[16][256];
  __shared__ float e2s[16][128];
  __shared__ float probv[16], invv[16];
  for (int i = lane; i < 16 * 256; i += 32) {
    const int m = i >> 8, k = i & 255;
    const float v = (k < 128) ? emb_e[(size_t)(n0 + m) * EMB + k]
                              : e2nl[((size_t)l * NE + n0 + m) * EMB + (k - 128)];
    X[m][k] = (_Float16)v;
  }
  if (lane < 16) {
    const int gn = n0 + lane;
    const float e2m = e2_multi[(size_t)l * NE + gn];
    const float p   = pred[(size_t)l * NE + gn];
    const float p2  = p / (S[l] + 1e-8f);
    const float rnd = u01((unsigned)l, (unsigned)gn, 0x9E3779B9u);
    probv[lane] = e2m * (1.0f / (p2 + 1e-8f)) * rnd;
    const float msk = bern05((unsigned)l, (unsigned)gn, 0x12345u);
    invv[lane] = (1.0f - e2m) * msk * 2.0f;
  }
  __syncthreads();
  const int hi = lane >> 4, lo = lane & 15;
  for (int nt = 0; nt < 8; ++nt) {
    const int f = nt * 16 + lo;
    CFrag g, cc;
    const float gb = g2_gb[f], cb = g2_bias[f];
    #pragma unroll
    for (int r = 0; r < 8; ++r) { g.f[r] = gb; cc.f[r] = cb; }
    #pragma unroll
    for (int kc = 0; kc < 8; ++kc) {
      HFrag a, bg, bc;
      const _Float16* ar = &X[lo][kc * 32 + hi * 8];
      a.q[0] = *(const f32x4*)ar;
      a.q[1] = *(const f32x4*)(ar + 16);
      const _Float16* bgp = WgT + (size_t)(kc * 32 + hi * 8) * EMB + nt * 16 + lo;
      const _Float16* bcp = WcT + (size_t)(kc * 32 + hi * 8) * EMB + nt * 16 + lo;
      #pragma unroll
      for (int j = 0; j < 8; ++j) {
        bg.h[j] = bgp[j * EMB]; bg.h[8 + j] = bgp[(16 + j) * EMB];
        bc.h[j] = bcp[j * EMB]; bc.h[8 + j] = bcp[(16 + j) * EMB];
      }
      g.v  = __builtin_amdgcn_wmma_f32_16x16x32_f16(false, a.v, false, bg.v,
                                                    (short)0, g.v, false, false);
      cc.v = __builtin_amdgcn_wmma_f32_16x16x32_f16(false, a.v, false, bc.v,
                                                    (short)0, cc.v, false, false);
    }
    #pragma unroll
    for (int r = 0; r < 8; ++r) {
      const int m = r + 8 * hi;
      const float ge = tanhf(g.f[r]);
      const float gt = 1.0f / (1.0f + __expf(-cc.f[r]));
      const float xe = (float)X[m][f];
      e2s[m][f] = (1.0f - gt) * xe + gt * ge;
    }
  }
  __syncthreads();
  const size_t pbase = (size_t)(l * NTILES + tile) * EMB;
  for (int f = lane; f < 128; f += 32) {
    float pp = 0.f, nn = 0.f;
    for (int m = 0; m < 16; ++m) {
      const float v = e2s[m][f];
      pp += v * probv[m];
      nn += v * invv[m];
    }
    posP[pbase + f] = pp; negP[pbase + f] = nn;
  }
  if (lane == 0) {
    float pw = 0.f, nw = 0.f;
    for (int m = 0; m < 16; ++m) { pw += probv[m]; nw += invv[m]; }
    posWP[l * NTILES + tile] = pw;
    negWP[l * NTILES + tile] = nw;
  }
}

// ---------------- K4: deterministic reduction + loss -------------------------
__global__ void __launch_bounds__(128) k4_finalize(
    const float* __restrict__ posP, const float* __restrict__ negP,
    const float* __restrict__ posWP, const float* __restrict__ negWP,
    const float* __restrict__ wse1g, const float* __restrict__ wstvec,
    float* __restrict__ loss_out)
{
  const int f = threadIdx.x;
  __shared__ float red[128];
  __shared__ float pwS[NB], nwS[NB], psS[NB], nsS[NB];
  if (f < NB) {
    float pw = 0.f, nw = 0.f;
    for (int t = 0; t < NTILES; ++t) { pw += posWP[f * NTILES + t]; nw += negWP[f * NTILES + t]; }
    pwS[f] = pw; nwS[f] = nw;
  }
  __syncthreads();
  for (int l = 0; l < NB; ++l) {
    float pa = 0.f, na = 0.f;
    for (int t = 0; t < NTILES; ++t) {
      const size_t base = (size_t)(l * NTILES + t) * EMB + f;
      pa += posP[base]; na += negP[base];
    }
    const float alpha = u01(0xA1FAu, (unsigned)l, 0x777u);
    const float beta  = u01(0xBE7Au, (unsigned)l, 0x555u);
    const float e1v = wse1g[l * EMB + f];
    float pm = pa / (pwS[l] + 1e-8f); pm = alpha * pm + (1.0f - alpha) * e1v;
    float nm = na / (nwS[l] + 1e-8f); nm = beta  * nm + (1.0f - beta)  * e1v;
    const float tv = wstvec[l * EMB + f];
    red[f] = fabsf(tv - pm);
    __syncthreads();
    for (int st = 64; st > 0; st >>= 1) { if (f < st) red[f] += red[f + st]; __syncthreads(); }
    if (f == 0) psS[l] = red[0];
    __syncthreads();
    red[f] = fabsf(tv - nm);
    __syncthreads();
    for (int st = 64; st > 0; st >>= 1) { if (f < st) red[f] += red[f + st]; __syncthreads(); }
    if (f == 0) nsS[l] = red[0];
    __syncthreads();
  }
  if (f == 0) {
    float s = 0.f;
    for (int l = 0; l < NB; ++l) {
      const float x = nsS[l] - psS[l];
      s += (x > 0.f) ? log1pf(__expf(-x)) : (-x + log1pf(__expf(x)));  // -log_sigmoid
    }
    loss_out[0] = s * (1.0f / NB);
  }
}

// ---------------- host launcher ---------------------------------------------
extern "C" void kernel_launch(void* const* d_in, const int* in_sizes, int n_in,
                              void* d_out, int out_size, void* d_ws, size_t ws_size,
                              hipStream_t stream)
{
  (void)in_sizes; (void)n_in; (void)out_size; (void)ws_size;
  const float* emb_e    = (const float*)d_in[0];
  const float* emb_rel  = (const float*)d_in[1];
  const float* emb_att  = (const float*)d_in[2];
  const float* lits     = (const float*)d_in[3];
  const float* lin_w    = (const float*)d_in[4];
  const float* lin_b    = (const float*)d_in[5];
  const float* in_w     = (const float*)d_in[6];
  const float* in_b     = (const float*)d_in[7];
  const float* out_w    = (const float*)d_in[8];
  const float* out_b    = (const float*)d_in[9];
  const float* g1_gw    = (const float*)d_in[10];
  const float* g1_gb    = (const float*)d_in[11];
  const float* g1_g1w   = (const float*)d_in[12];
  const float* g1_g2w   = (const float*)d_in[13];
  const float* g1_bias  = (const float*)d_in[14];
  const float* g2_gw    = (const float*)d_in[15];
  const float* g2_gb    = (const float*)d_in[16];
  const float* g2_g1w   = (const float*)d_in[17];
  const float* g2_g2w   = (const float*)d_in[18];
  const float* g2_bias  = (const float*)d_in[19];
  const int*   e1       = (const int*)d_in[20];
  const int*   rel      = (const int*)d_in[21];
  const float* e2_multi = (const float*)d_in[23];

  float* pred_out = (float*)d_out;            // 8 x 10000
  float* loss_out = pred_out + (size_t)NB * NE;

  float* p = (float*)d_ws;
  float* wsA     = p; p += 4096;
  float* wsc     = p; p += 64;
  float* wsbias2 = p; p += 128;
  float* wstvec  = p; p += NB * EMB;
  float* wse1g   = p; p += NB * EMB;
  float* Spart   = p; p += NB * NTILES;
  float* S       = p; p += NB;
  float* posP    = p; p += (size_t)NB * NTILES * EMB;
  float* negP    = p; p += (size_t)NB * NTILES * EMB;
  float* posWP   = p; p += NB * NTILES;
  float* negWP   = p; p += NB * NTILES;
  _Float16* U16  = (_Float16*)p;
  _Float16* WgT  = U16 + 512 * 128;
  _Float16* WcT  = WgT + 256 * 128;
  float* e2nl = (float*)(((uintptr_t)(WcT + 256 * 128) + 255) & ~(uintptr_t)255); // 8*10000*128 f32

  k0_precompute<<<1, 128, 0, stream>>>(emb_e, emb_rel, emb_att, lits, lin_w, lin_b,
      in_w, in_b, out_w, out_b, g1_gw, g1_gb, g1_g1w, g1_g2w, g1_bias,
      g2_gw, g2_g1w, g2_g2w, e1, rel,
      wsA, wsc, wsbias2, wstvec, wse1g, U16, WgT, WcT);

  dim3 grid(NTILES, NB);
  k1_attention<<<grid, 32, 0, stream>>>(lits, wsA, wsc, wsbias2, U16, e2nl);
  k2_gate_score<<<grid, 32, 0, stream>>>(emb_e, e2nl, WgT, WcT, g2_gb, g2_bias,
                                         wstvec, e2_multi, pred_out, Spart);
  k2b_reduceS<<<1, 32, 0, stream>>>(Spart, S);
  k3_loss_accum<<<grid, 32, 0, stream>>>(emb_e, e2nl, WgT, WcT, g2_gb, g2_bias,
                                         e2_multi, pred_out, S, posP, negP, posWP, negWP);
  k4_finalize<<<1, 128, 0, stream>>>(posP, negP, posWP, negWP, wse1g, wstvec, loss_out);
}
